// MultiHeadSelfAttention_26628797235811
// MI455X (gfx1250) — matrix operand order repaired and checked by reading, not compiled, not run on hardware
//
#include <hip/hip_runtime.h>
#include <hip/hip_bf16.h>

// Problem constants (from reference)
#define B_   4
#define C_   128
#define T_   512
#define F_   128
#define H_   4
#define D_   64
#define DV_  32
#define DF_  (D_ * F_)    // 8192
#define DVF_ (DV_ * F_)   // 4096
#define EPS_ 1e-5f

typedef __bf16 bf16_t;
typedef __attribute__((ext_vector_type(16))) __bf16         v16bf;
typedef __attribute__((ext_vector_type(8)))  __bf16         v8bf;
typedef __attribute__((ext_vector_type(8)))  float          v8f;
typedef __attribute__((ext_vector_type(8)))  unsigned short us8;

__device__ __forceinline__ unsigned short f2bf(float f) {
  return __builtin_bit_cast(unsigned short, (__bf16)f);   // hardware cvt, RNE
}
__device__ __forceinline__ v8bf ld8(const unsigned short* p) {   // 16B vector load
  return __builtin_bit_cast(v8bf, *(const us8*)p);
}
__device__ __forceinline__ v16bf cat(v8bf lo, v8bf hi) {
  return __builtin_shufflevector(lo, hi, 0,1,2,3,4,5,6,7,8,9,10,11,12,13,14,15);
}
__device__ __forceinline__ v8f wmma_bf16(v16bf a, v16bf b, v8f c) {
  return __builtin_amdgcn_wmma_f32_16x16x32_bf16(false, a, false, b, (short)0, c, false, false);
}

// ---------------------------------------------------------------------------
// K0: one-time weight conversion to bf16.
// Wbf: stacked per-head rows [H][160][128]  (0-63 q, 64-127 k, 128-159 v)
// Wobf: [128][128]
// ---------------------------------------------------------------------------
__global__ __launch_bounds__(256) void k_prep_w(
    const float* __restrict__ Wq, const float* __restrict__ Wk,
    const float* __restrict__ Wv, const float* __restrict__ Wo,
    unsigned short* __restrict__ Wbf, unsigned short* __restrict__ Wobf)
{
  const int i = blockIdx.x * 256 + threadIdx.x;
  if (i < H_ * 160 * C_) {
    const int c = i & 127;
    const int rowg = i >> 7;
    const int h = rowg / 160, row = rowg % 160;
    float w;
    if (row < 64)       w = Wq[(h * 64 + row) * C_ + c];
    else if (row < 128) w = Wk[(h * 64 + (row - 64)) * C_ + c];
    else                w = Wv[(h * 32 + (row - 128)) * C_ + c];
    Wbf[i] = f2bf(w);
  }
  if (i < C_ * C_) Wobf[i] = f2bf(Wo[i]);
}

// ---------------------------------------------------------------------------
// K1: fused QKV projection + bias + PReLU + LayerNorm.  One WG per (b,h,t).
// Y[160x128] = Wstack[160x128] @ X[128x128]
// ---------------------------------------------------------------------------
__global__ __launch_bounds__(256) void k_proj_ln(
    const float* __restrict__ x, const unsigned short* __restrict__ Wbf,
    const float* __restrict__ bq, const float* __restrict__ aq,
    const float* __restrict__ lwq, const float* __restrict__ lbq,
    const float* __restrict__ bk, const float* __restrict__ ak,
    const float* __restrict__ lwk, const float* __restrict__ lbk,
    const float* __restrict__ bv, const float* __restrict__ av,
    const float* __restrict__ lwv, const float* __restrict__ lbv,
    unsigned short* __restrict__ qf, unsigned short* __restrict__ kf,
    unsigned short* __restrict__ vf)
{
  const int LP = 136;                      // padded transposed row stride (16B multiple)
  __shared__ unsigned short Xt[F_ * LP];   // Xt[f][c] bf16 (transposed x tile)
  __shared__ float red[6];

  const int blk = blockIdx.x;
  const int t = blk % T_;
  const int h = (blk / T_) % H_;
  const int b = blk / (T_ * H_);
  const int tid = threadIdx.x;
  if (tid < 6) red[tid] = 0.f;

  // coalesced global read (consecutive f), transposed LDS write
  for (int i = tid; i < C_ * F_; i += 256) {
    int c = i >> 7, f = i & 127;
    Xt[f * LP + c] = f2bf(x[((size_t)(b * C_ + c) * T_ + t) * F_ + f]);
  }
  __syncthreads();

  const int lane = tid & 31, wave = tid >> 5;
  const int hf = lane >> 4, ln = lane & 15;
  const int f0 = wave * 16;

  // B fragments: lane -> column f0+ln; elements in table K(=c) order
  v16bf Bf[4];
#pragma unroll
  for (int s = 0; s < 4; ++s) {
    const unsigned short* p = &Xt[(f0 + ln) * LP + s * 32 + 8 * hf];
    Bf[s] = cat(ld8(p), ld8(p + 16));
  }

  v8f acc[10];
#pragma unroll
  for (int mt = 0; mt < 10; ++mt) {
    v8f a = {};
    const unsigned short* Wrow = Wbf + (size_t)(h * 160 + mt * 16 + ln) * C_;
#pragma unroll
    for (int s = 0; s < 4; ++s) {
      const unsigned short* p = Wrow + s * 32 + 8 * hf;
      a = wmma_bf16(cat(ld8(p), ld8(p + 16)), Bf[s], a);
    }
    acc[mt] = a;
  }

  // bias + PReLU in place, LayerNorm partials per region (q/k/v)
  const float aqh = aq[h], akh = ak[h], avh = av[h];
  float ps[3] = {0.f, 0.f, 0.f}, pq[3] = {0.f, 0.f, 0.f};
#pragma unroll
  for (int mt = 0; mt < 10; ++mt) {
    const int rg = (mt < 4) ? 0 : (mt < 8 ? 1 : 2);
    const float al = (rg == 0) ? aqh : (rg == 1 ? akh : avh);
#pragma unroll
    for (int e = 0; e < 8; ++e) {
      const int m = mt * 16 + e + 8 * hf;
      float bias;
      if (rg == 0)      bias = bq[h * 64 + m];
      else if (rg == 1) bias = bk[h * 64 + (m - 64)];
      else              bias = bv[h * 32 + (m - 128)];
      float y = acc[mt][e] + bias;
      y = (y >= 0.f) ? y : al * y;
      acc[mt][e] = y;
      ps[rg] += y; pq[rg] += y * y;
    }
  }
  atomicAdd(&red[0], ps[0]); atomicAdd(&red[1], ps[1]); atomicAdd(&red[2], ps[2]);
  atomicAdd(&red[3], pq[0]); atomicAdd(&red[4], pq[1]); atomicAdd(&red[5], pq[2]);
  __syncthreads();

  float mu[3], rs[3];
#pragma unroll
  for (int rg = 0; rg < 3; ++rg) {
    const float inv = (rg == 2) ? (1.f / 4096.f) : (1.f / 8192.f);
    mu[rg] = red[rg] * inv;
    float var = red[rg + 3] * inv - mu[rg] * mu[rg];
    rs[rg] = rsqrtf(var + EPS_);
  }

  const int bh = b * H_ + h;
#pragma unroll
  for (int mt = 0; mt < 10; ++mt) {
    const int rg = (mt < 4) ? 0 : (mt < 8 ? 1 : 2);
#pragma unroll
    for (int e = 0; e < 8; ++e) {
      const int m = mt * 16 + e + 8 * hf;
      const int f = f0 + ln;
      const float yn = (acc[mt][e] - mu[rg]) * rs[rg];
      if (rg == 0) {
        const int d = m;
        const float o = yn * lwq[(h * 64 + d) * F_ + f] + lbq[(h * 64 + d) * F_ + f];
        qf[(size_t)(bh * T_ + t) * DF_ + d * F_ + f] = f2bf(o);
      } else if (rg == 1) {
        const int d = m - 64;
        const float o = yn * lwk[(h * 64 + d) * F_ + f] + lbk[(h * 64 + d) * F_ + f];
        kf[(size_t)(bh * T_ + t) * DF_ + d * F_ + f] = f2bf(o);
      } else {
        const int d = m - 128;
        const float o = yn * lwv[(h * 32 + d) * F_ + f] + lbv[(h * 32 + d) * F_ + f];
        vf[(size_t)(bh * T_ + t) * DVF_ + d * F_ + f] = f2bf(o);
      }
    }
  }
}

// ---------------------------------------------------------------------------
// K2: S = scale * qf @ kf^T per (b,h).  LDS-tiled GEMM: WG computes a 128x128
// tile of S; stages 128x64 K-slabs of qf and kf in LDS (36 KB); each of the
// 8 waves owns a 16-row strip x 8 column tiles (64 acc VGPRs/lane).
// Cuts global/L2 traffic ~8x vs wave-per-tile streaming.
// ---------------------------------------------------------------------------
__global__ __launch_bounds__(256) void k_scores(
    const unsigned short* __restrict__ qf, const unsigned short* __restrict__ kf,
    float* __restrict__ S)
{
  const int LP = 72;                          // padded 64-K row stride (16B multiple)
  __shared__ unsigned short ABs[2 * 128 * 72];  // A slab | B slab, 36 KB
  unsigned short* As = ABs;
  unsigned short* Bs = ABs + 128 * 72;

  const int blk = blockIdx.x;                 // 16 bh * 4 * 4
  const int nt0 = (blk & 3) * 8;              // column tile base (16-wide tiles)
  const int mt0 = ((blk >> 2) & 3) * 8;       // row tile base
  const int bh  = blk >> 4;

  const int tid = threadIdx.x;
  const int lane = tid & 31, wave = tid >> 5;
  const int hf = lane >> 4, ln = lane & 15;

  const unsigned short* qbase = qf + (size_t)bh * T_ * DF_ + (size_t)(mt0 * 16) * DF_;
  const unsigned short* kbase = kf + (size_t)bh * T_ * DF_ + (size_t)(nt0 * 16) * DF_;

  v8f acc[8] = {};

  for (int kk = 0; kk < DF_; kk += 64) {
    // stage both slabs: 2 slabs * 128 rows * 8 x 16B chunks = 8 chunks/thread
    for (int i = tid; i < 2048; i += 256) {
      const int slab = i >> 10;
      const int row  = (i >> 3) & 127;
      const int ch   = i & 7;
      const unsigned short* src = (slab ? kbase : qbase) + (size_t)row * DF_ + kk + ch * 8;
      *(us8*)((slab ? Bs : As) + row * LP + ch * 8) = *(const us8*)src;
    }
    __syncthreads();

#pragma unroll
    for (int s = 0; s < 2; ++s) {             // two 32-K WMMA steps per slab
      const unsigned short* pa = As + (wave * 16 + ln) * LP + s * 32 + 8 * hf;
      v16bf A = cat(ld8(pa), ld8(pa + 16));
#pragma unroll
      for (int nt = 0; nt < 8; ++nt) {
        const unsigned short* pb = Bs + (nt * 16 + ln) * LP + s * 32 + 8 * hf;
        acc[nt] = wmma_bf16(A, cat(ld8(pb), ld8(pb + 16)), acc[nt]);
      }
    }
    __syncthreads();
  }

  const float scale = 0.011048543456039806f;  // 1/sqrt(D*F)
  float* Srow = S + (size_t)bh * T_ * T_;
#pragma unroll
  for (int nt = 0; nt < 8; ++nt)
#pragma unroll
    for (int e = 0; e < 8; ++e)
      Srow[(size_t)((mt0 + wave) * 16 + e + 8 * hf) * T_ + (nt0 + nt) * 16 + ln]
          = acc[nt][e] * scale;
}

// ---------------------------------------------------------------------------
// K3: row softmax over 512 scores; writes bf16 probabilities.
// ---------------------------------------------------------------------------
__global__ __launch_bounds__(256) void k_softmax(
    const float* __restrict__ S, unsigned short* __restrict__ P)
{
  __shared__ float red[256];
  const int row = blockIdx.x;
  const int tid = threadIdx.x;
  const float* s = S + (size_t)row * T_;
  const float v0 = s[tid], v1 = s[tid + 256];

  red[tid] = fmaxf(v0, v1); __syncthreads();
  for (int off = 128; off > 0; off >>= 1) {
    if (tid < off) red[tid] = fmaxf(red[tid], red[tid + off]);
    __syncthreads();
  }
  const float mx = red[0]; __syncthreads();

  const float e0 = __expf(v0 - mx), e1 = __expf(v1 - mx);
  red[tid] = e0 + e1; __syncthreads();
  for (int off = 128; off > 0; off >>= 1) {
    if (tid < off) red[tid] += red[tid + off];
    __syncthreads();
  }
  const float inv = 1.f / red[0];
  unsigned short* p = P + (size_t)row * T_;
  p[tid] = f2bf(e0 * inv);
  p[tid + 256] = f2bf(e1 * inv);
}

// ---------------------------------------------------------------------------
// K4: out = P @ vf per (b,h).  WG = (bh, mt, 128-wide N group); vf K-slab is
// cooperatively staged transposed in LDS so B-fragments are 2x ds_load_b128.
// ---------------------------------------------------------------------------
__global__ __launch_bounds__(256) void k_av(
    const unsigned short* __restrict__ P, const unsigned short* __restrict__ vf,
    unsigned short* __restrict__ r)
{
  const int LPK = 40;                      // padded K stride (16B multiple)
  __shared__ unsigned short Vt[128 * LPK]; // Vt[nloc][kloc], 10 KB

  const int blk = blockIdx.x;              // 16 * 32 * 32
  const int ng = blk & 31;                 // N group (128 cols)
  const int mt = (blk >> 5) & 31;
  const int bh = blk >> 10;
  const int b = bh >> 2, h = bh & 3;
  const int n0 = ng * 128;

  const int tid = threadIdx.x;
  const int lane = tid & 31, wave = tid >> 5;
  const int hf = lane >> 4, ln = lane & 15;

  const unsigned short* prow = P + (size_t)bh * T_ * T_ + (size_t)(mt * 16 + ln) * T_;
  const unsigned short* vbase = vf + (size_t)bh * T_ * DVF_ + n0;

  v8f acc = {};
  for (int kk = 0; kk < T_; kk += 32) {
    // stage vf[kk..kk+32)[n0..n0+128) transposed (coalesced over n)
    for (int i = tid; i < 32 * 128; i += 256) {
      int kloc = i >> 7, nloc = i & 127;
      Vt[nloc * LPK + kloc] = vbase[(size_t)(kk + kloc) * DVF_ + nloc];
    }
    __syncthreads();

    v16bf A = cat(ld8(prow + kk + 8 * hf), ld8(prow + kk + 16 + 8 * hf));
    const unsigned short* p = &Vt[(wave * 16 + ln) * LPK + 8 * hf];
    v16bf Bv = cat(ld8(p), ld8(p + 16));
    acc = wmma_bf16(A, Bv, acc);
    __syncthreads();
  }

#pragma unroll
  for (int e = 0; e < 8; ++e) {
    const int t = mt * 16 + e + 8 * hf;
    const int col = n0 + wave * 16 + ln;
    r[(size_t)(b * T_ + t) * (C_ * F_) + h * DVF_ + col] = f2bf(acc[e]);
  }
}

// ---------------------------------------------------------------------------
// K5: z = LN(PReLU(Wo @ r + bo)) * lwo + lbo + x.  One WG per (b,t).
// ---------------------------------------------------------------------------
__global__ __launch_bounds__(256) void k_out(
    const unsigned short* __restrict__ r, const float* __restrict__ x,
    const unsigned short* __restrict__ Wobf, const float* __restrict__ bo,
    const float* __restrict__ ao,
    const float* __restrict__ lwo, const float* __restrict__ lbo,
    float* __restrict__ out)
{
  const int LP = 136;
  __shared__ unsigned short Rt[F_ * LP];   // Rt[f][c] (transposed r tile)
  __shared__ float red[2];
  const int blk = blockIdx.x;
  const int t = blk % T_, b = blk / T_;
  const int tid = threadIdx.x;
  if (tid < 2) red[tid] = 0.f;

  for (int i = tid; i < C_ * F_; i += 256) {
    int c = i >> 7, f = i & 127;
    Rt[f * LP + c] = r[(size_t)(b * T_ + t) * (C_ * F_) + i];
  }
  __syncthreads();

  const int lane = tid & 31, wave = tid >> 5;
  const int hf = lane >> 4, ln = lane & 15;
  const int f0 = wave * 16;

  v16bf Bf[4];
#pragma unroll
  for (int s = 0; s < 4; ++s) {
    const unsigned short* p = &Rt[(f0 + ln) * LP + s * 32 + 8 * hf];
    Bf[s] = cat(ld8(p), ld8(p + 16));
  }

  v8f acc[8];
#pragma unroll
  for (int mt = 0; mt < 8; ++mt) {
    v8f a = {};
    const unsigned short* Wrow = Wobf + (size_t)(mt * 16 + ln) * C_;
#pragma unroll
    for (int s = 0; s < 4; ++s) {
      const unsigned short* p = Wrow + s * 32 + 8 * hf;
      a = wmma_bf16(cat(ld8(p), ld8(p + 16)), Bf[s], a);
    }
    acc[mt] = a;
  }

  const float aoi = ao[0];
  float ps = 0.f, pq = 0.f;
#pragma unroll
  for (int mt = 0; mt < 8; ++mt)
#pragma unroll
    for (int e = 0; e < 8; ++e) {
      const int d = mt * 16 + e + 8 * hf;
      float y = acc[mt][e] + bo[d];
      y = (y >= 0.f) ? y : aoi * y;
      acc[mt][e] = y;
      ps += y; pq += y * y;
    }
  atomicAdd(&red[0], ps); atomicAdd(&red[1], pq);
  __syncthreads();
  const float mu = red[0] * (1.f / 16384.f);
  const float var = red[1] * (1.f / 16384.f) - mu * mu;
  const float rstd = rsqrtf(var + EPS_);

#pragma unroll
  for (int mt = 0; mt < 8; ++mt)
#pragma unroll
    for (int e = 0; e < 8; ++e) {
      const int d = mt * 16 + e + 8 * hf;
      const int f = f0 + ln;
      const float yn = (acc[mt][e] - mu) * rstd;
      const size_t xi = ((size_t)(b * C_ + d) * T_ + t) * F_ + f;
      out[xi] = yn * lwo[d * F_ + f] + lbo[d * F_ + f] + x[xi];
    }
}

// ---------------------------------------------------------------------------
extern "C" void kernel_launch(void* const* d_in, const int* in_sizes, int n_in,
                              void* d_out, int out_size, void* d_ws, size_t ws_size,
                              hipStream_t stream) {
  const float* x   = (const float*)d_in[0];
  const float* Wq  = (const float*)d_in[1];
  const float* bq  = (const float*)d_in[2];
  const float* aq  = (const float*)d_in[3];
  const float* lwq = (const float*)d_in[4];
  const float* lbq = (const float*)d_in[5];
  const float* Wk  = (const float*)d_in[6];
  const float* bk  = (const float*)d_in[7];
  const float* ak  = (const float*)d_in[8];
  const float* lwk = (const float*)d_in[9];
  const float* lbk = (const float*)d_in[10];
  const float* Wv  = (const float*)d_in[11];
  const float* bv  = (const float*)d_in[12];
  const float* av  = (const float*)d_in[13];
  const float* lwv = (const float*)d_in[14];
  const float* lbv = (const float*)d_in[15];
  const float* Wo  = (const float*)d_in[16];
  const float* bo  = (const float*)d_in[17];
  const float* ao  = (const float*)d_in[18];
  const float* lwo = (const float*)d_in[19];
  const float* lbo = (const float*)d_in[20];

  char* ws = (char*)d_ws;
  unsigned short* qf   = (unsigned short*)ws; ws += (size_t)B_ * H_ * T_ * DF_ * 2;
  unsigned short* kf   = (unsigned short*)ws; ws += (size_t)B_ * H_ * T_ * DF_ * 2;
  unsigned short* vf   = (unsigned short*)ws; ws += (size_t)B_ * H_ * T_ * DVF_ * 2;
  float*          S    = (float*)ws;          ws += (size_t)B_ * H_ * T_ * T_ * 4;
  unsigned short* P    = (unsigned short*)ws; ws += (size_t)B_ * H_ * T_ * T_ * 2;
  unsigned short* R    = (unsigned short*)ws; ws += (size_t)B_ * T_ * C_ * F_ * 2;
  unsigned short* Wbf  = (unsigned short*)ws; ws += (size_t)H_ * 160 * C_ * 2;
  unsigned short* Wobf = (unsigned short*)ws;

  k_prep_w<<<(H_ * 160 * C_ + 255) / 256, 256, 0, stream>>>(Wq, Wk, Wv, Wo, Wbf, Wobf);
  k_proj_ln<<<B_ * H_ * T_, 256, 0, stream>>>(x, Wbf, bq, aq, lwq, lbq,
                                              bk, ak, lwk, lbk,
                                              bv, av, lwv, lbv, qf, kf, vf);
  k_scores<<<B_ * H_ * 4 * 4, 256, 0, stream>>>(qf, kf, S);
  k_softmax<<<B_ * H_ * T_, 256, 0, stream>>>(S, P);
  k_av<<<B_ * H_ * 32 * 32, 256, 0, stream>>>(P, vf, R);
  k_out<<<B_ * T_, 256, 0, stream>>>(R, x, Wobf, bo, ao, lwo, lbo, (float*)d_out);
}
